// HyFreqMoE_SR_773094113666
// MI455X (gfx1250) — compile-verified, matrix operand order checked by reading, not run on hardware
//
#include <hip/hip_runtime.h>

typedef __attribute__((ext_vector_type(16))) _Float16 v16h;
typedef __attribute__((ext_vector_type(8)))  float    v8f;

// ---------- constants ----------
#define DIMC   192
#define HIDC   384            // HID = 2*DIM -> split halves are 384 each
#define B2HID  768            // 2*HID
#define QKVC   576
#define HWSZ   65536          // 256*256
#define BATCH  4
#define MTOK   262144         // 4*256*256 tokens
#define PER_B  12582912       // 192*65536 elements per batch
#define INVPB  (1.0f/12582912.0f)

// workspace layout (bytes)
#define WS_MASTER 0ull                    // f32 [MTOK][192]        201326592
#define WS_NORMED 201326592ull            // f16 [MTOK][192]        100663296
#define WS_BIG    301989888ull            // f16 [MTOK][768]        402653184
#define WS_GATED  704643072ull            // f16 [MTOK][384]        201326592
#define WS_STATS  905969664ull            // f32 [3][4][2]
#define WS_WF16   905969920ull            // f16 weights pool

// f16 weight pool offsets (halves)
#define WF_PW1  0
#define WF_PW2  147456
#define WF_QKV  221184
#define WF_PROJ 331776
#define WF_FC1  368640
#define WF_FC2  516096

__device__ __forceinline__ float gelu_exact(float v) {
  return 0.5f * v * (1.0f + erff(v * 0.70710678118654752f));
}

// Load a 16x16x32 f16 WMMA operand fragment from a row-major [*,32] LDS row.
// ISA 7.12.2: lanes 0-15 -> K {0..7,16..23}; lanes 16-31 -> K {8..15,24..31}.
__device__ __forceinline__ v16h ld_frag_row(const _Float16* row, int kb) {
  v16h f;
#pragma unroll
  for (int j = 0; j < 8; ++j) { f[j] = row[kb + j]; f[j + 8] = row[kb + 16 + j]; }
  return f;
}

__device__ __forceinline__ v8f wmma_f16(v16h a, v16h b, v8f c) {
  return __builtin_amdgcn_wmma_f32_16x16x32_f16(false, a, false, b, (short)0, c,
                                                false, false);
}

// ---------------- init / transpose / convert ----------------
__global__ void k_init_stats(float* stats) {
  if (threadIdx.x < 24) stats[threadIdx.x] = 0.0f;
}

__global__ void k_f32_to_f16(const float* __restrict__ src,
                             _Float16* __restrict__ dst, int n) {
  int i = blockIdx.x * 256 + threadIdx.x;
  if (i < n) dst[i] = (_Float16)src[i];
}

__global__ void k_nchw_to_nhwc(const float* __restrict__ x,
                               float* __restrict__ master) {
  size_t i = (size_t)blockIdx.x * 256 + threadIdx.x;   // over 50331648
  int b = (int)(i / PER_B);
  int r = (int)(i - (size_t)b * PER_B);
  int c = r >> 16;
  int hw = r & 65535;
  master[((size_t)b * HWSZ + hw) * DIMC + c] = x[i];
}

// ---------------- group norm ----------------
__global__ __launch_bounds__(256) void k_gn_stats(const float* __restrict__ x,
                                                  float* __restrict__ stats) {
  const int b = blockIdx.y;
  const float4* p = reinterpret_cast<const float4*>(x + (size_t)b * PER_B);
  float s = 0.f, q = 0.f;
  for (size_t i = (size_t)blockIdx.x * 256 + threadIdx.x; i < (PER_B / 4);
       i += (size_t)gridDim.x * 256) {
    float4 v = p[i];
    s += v.x + v.y + v.z + v.w;
    q += v.x * v.x + v.y * v.y + v.z * v.z + v.w * v.w;
  }
  __shared__ float rs[256], rq[256];
  int t = threadIdx.x;
  rs[t] = s; rq[t] = q; __syncthreads();
  for (int o = 128; o > 0; o >>= 1) {
    if (t < o) { rs[t] += rs[t + o]; rq[t] += rq[t + o]; }
    __syncthreads();
  }
  if (t == 0) {
    atomicAdd(&stats[b * 2 + 0], rs[0]);
    atomicAdd(&stats[b * 2 + 1], rq[0]);
  }
}

__global__ __launch_bounds__(256) void k_gn_apply(
    const float* __restrict__ x, const float* __restrict__ stats,
    const float* __restrict__ g, const float* __restrict__ bb,
    _Float16* __restrict__ out) {
  size_t i4 = (size_t)blockIdx.x * 256 + threadIdx.x;   // over 12582912
  size_t i = i4 * 4;
  int c = (int)(i % DIMC);
  int b = (int)(i / PER_B);
  float mean = stats[b * 2 + 0] * INVPB;
  float var  = stats[b * 2 + 1] * INVPB - mean * mean;
  float rs = rsqrtf(var + 1e-6f);
  float4 v = *reinterpret_cast<const float4*>(&x[i]);
  _Float16 o[4];
  o[0] = (_Float16)((v.x - mean) * rs * g[c + 0] + bb[c + 0]);
  o[1] = (_Float16)((v.y - mean) * rs * g[c + 1] + bb[c + 1]);
  o[2] = (_Float16)((v.z - mean) * rs * g[c + 2] + bb[c + 2]);
  o[3] = (_Float16)((v.w - mean) * rs * g[c + 3] + bb[c + 3]);
  *reinterpret_cast<uint2*>(&out[i]) = *reinterpret_cast<uint2*>(o);
}

// ---------------- WMMA GEMM: C[M,N] = A[M,K] * W[N,K]^T + bias ----------------
// Block tile: 64 rows x (NACC*32) cols. 8 waves = 4 m-tiles x 2 n-groups; each
// wave owns NACC 16x16 f32 accumulators -> NACC WMMAs per (NACC+1) LDS frags.
// MODE 0: store f16 out    MODE 1: master += scale*val
// MODE 2: out_nchw = master + scale*val
template <int MODE, int NACC>
__global__ __launch_bounds__(256) void k_gemm(
    const _Float16* __restrict__ A, const _Float16* __restrict__ W,
    const float* __restrict__ bias, int N, int K,
    _Float16* __restrict__ outh, const float* __restrict__ scale_ptr,
    float* __restrict__ master, float* __restrict__ out_nchw) {
  constexpr int BN = NACC * 32;           // block n-extent
  __shared__ _Float16 As[64 * 32];
  __shared__ _Float16 Bs[BN * 32];
  const int m0 = blockIdx.x * 64;
  const int n0 = blockIdx.y * BN;
  const int t = threadIdx.x;
  const int wv = t >> 5, lane = t & 31;
  const int mt = wv & 3, ng = wv >> 2;
  const int kb = (lane < 16) ? 0 : 8;
  const int ln = lane & 15;
  v8f acc[NACC];
#pragma unroll
  for (int u = 0; u < NACC; ++u) acc[u] = v8f{};

  for (int k0 = 0; k0 < K; k0 += 32) {
    __syncthreads();
    {  // A tile: 64x32 halves = 256 uint4 chunks (one per thread)
      const int row = t >> 2, part = t & 3;
      *reinterpret_cast<uint4*>(&As[t * 8]) = *reinterpret_cast<const uint4*>(
          &A[(size_t)(m0 + row) * K + k0 + part * 8]);
      if (k0 + 32 < K)
        __builtin_prefetch(&A[(size_t)(m0 + row) * K + k0 + 32 + part * 8], 0, 3);
    }
#pragma unroll
    for (int c = t; c < BN * 4; c += 256) {  // B tile: BN x 32 halves
      const int row = c >> 2, part = c & 3;
      *reinterpret_cast<uint4*>(&Bs[c * 8]) = *reinterpret_cast<const uint4*>(
          &W[(size_t)(n0 + row) * K + k0 + part * 8]);
      if (k0 + 32 < K)
        __builtin_prefetch(&W[(size_t)(n0 + row) * K + k0 + 32 + part * 8], 0, 3);
    }
    __syncthreads();
    v16h af = ld_frag_row(&As[(mt * 16 + ln) * 32], kb);
#pragma unroll
    for (int u = 0; u < NACC; ++u) {
      v16h bf = ld_frag_row(&Bs[(ng * (NACC * 16) + u * 16 + ln) * 32], kb);
      acc[u] = wmma_f16(af, bf, acc[u]);
    }
  }

  float scale = 0.0f;
  if constexpr (MODE != 0) scale = *scale_ptr;
#pragma unroll
  for (int u = 0; u < NACC; ++u) {
    const int n_g = n0 + ng * (NACC * 16) + u * 16 + ln;
    const float bv = bias[n_g];
#pragma unroll
    for (int r = 0; r < 8; ++r) {
      const int m = mt * 16 + ((lane < 16) ? r : 8 + r);
      const size_t tok = (size_t)m0 + m;
      const float val = acc[u][r] + bv;
      if constexpr (MODE == 0) {
        outh[tok * (size_t)N + n_g] = (_Float16)val;
      } else if constexpr (MODE == 1) {
        const size_t i = tok * DIMC + n_g;
        master[i] = master[i] + scale * val;
      } else {
        const size_t i = tok * DIMC + n_g;
        const int bi = (int)(tok >> 16);
        const int hw = (int)(tok & 65535);
        out_nchw[(((size_t)bi * DIMC + n_g) << 16) + hw] =
            master[i] + scale * val;
      }
    }
  }
}

// ---------------- depthwise 3x3 + gated GELU ----------------
__global__ __launch_bounds__(256) void k_dwconv_gate(
    const _Float16* __restrict__ y, const float* __restrict__ dw_w,
    const float* __restrict__ dw_b, _Float16* __restrict__ gated) {
  size_t idx = (size_t)blockIdx.x * 256 + threadIdx.x;   // over MTOK*48
  size_t tok = idx / 48;
  int c0 = (int)(idx % 48) * 8;
  int b = (int)(tok >> 16);
  int hw = (int)(tok & 65535);
  int h = hw >> 8, w = hw & 255;
  float au[8] = {}, av[8] = {};
#pragma unroll
  for (int dh = -1; dh <= 1; ++dh) {
#pragma unroll
    for (int dw = -1; dw <= 1; ++dw) {
      int hh = h + dh, ww = w + dw;
      if (hh < 0 || hh > 255 || ww < 0 || ww > 255) continue;
      const _Float16* row =
          y + ((size_t)b * HWSZ + (hh << 8) + ww) * B2HID;
      uint4 uu = *reinterpret_cast<const uint4*>(&row[c0]);
      uint4 vv = *reinterpret_cast<const uint4*>(&row[c0 + HIDC]);
      const _Float16* up = reinterpret_cast<const _Float16*>(&uu);
      const _Float16* vp = reinterpret_cast<const _Float16*>(&vv);
      int tap = (dh + 1) * 3 + (dw + 1);
#pragma unroll
      for (int j = 0; j < 8; ++j) {
        au[j] += (float)up[j] * dw_w[(c0 + j) * 9 + tap];
        av[j] += (float)vp[j] * dw_w[(c0 + j + HIDC) * 9 + tap];
      }
    }
  }
  _Float16 o[8];
#pragma unroll
  for (int j = 0; j < 8; ++j) {
    float uf = au[j] + dw_b[c0 + j];
    float vf = av[j] + dw_b[c0 + j + HIDC];
    o[j] = (_Float16)(uf * gelu_exact(vf));
  }
  *reinterpret_cast<uint4*>(&gated[tok * HIDC + c0]) =
      *reinterpret_cast<uint4*>(o);
}

// ---------------- FFN gate (no conv) ----------------
__global__ __launch_bounds__(256) void k_gate(const _Float16* __restrict__ y,
                                              _Float16* __restrict__ gated) {
  size_t idx = (size_t)blockIdx.x * 256 + threadIdx.x;   // over MTOK*48
  size_t tok = idx / 48;
  int c0 = (int)(idx % 48) * 8;
  const _Float16* row = y + tok * B2HID;
  uint4 uu = *reinterpret_cast<const uint4*>(&row[c0]);
  uint4 vv = *reinterpret_cast<const uint4*>(&row[c0 + HIDC]);
  const _Float16* up = reinterpret_cast<const _Float16*>(&uu);
  const _Float16* vp = reinterpret_cast<const _Float16*>(&vv);
  _Float16 o[8];
#pragma unroll
  for (int j = 0; j < 8; ++j)
    o[j] = (_Float16)((float)up[j] * gelu_exact((float)vp[j]));
  *reinterpret_cast<uint4*>(&gated[tok * HIDC + c0]) =
      *reinterpret_cast<uint4*>(o);
}

// ---------------- shifted window attention (per window, per head) ----------
__global__ __launch_bounds__(128) void k_attn(const _Float16* __restrict__ qkv,
                                              _Float16* __restrict__ obuf) {
  __shared__ _Float16 qS[64 * 32];
  __shared__ _Float16 kS[64 * 32];
  __shared__ _Float16 vS[64 * 32];
  __shared__ float    sS[64 * 64];
  __shared__ _Float16 pS[64 * 64];

  const int win = blockIdx.x, head = blockIdx.y;
  const int b = win >> 10;
  const int rem = win & 1023;
  const int wh = rem >> 5, ww = rem & 31;
  const int t = threadIdx.x;
  const int wv = t >> 5, lane = t & 31;
  const int kb = (lane < 16) ? 0 : 8;
  const int ln = lane & 15;

  auto tok_g = [&](int i) -> size_t {
    int ti = i >> 3, tj = i & 7;
    int h = (wh * 8 + ti + 4) & 255;      // fold roll(-4) gather / roll(+4) scatter
    int w = (ww * 8 + tj + 4) & 255;
    return ((size_t)b << 16) + (h << 8) + w;
  };

  // gather q,k,v for this head: 64 tokens x 3 mats x 4 uint4-chunks
  for (int c = t; c < 768; c += 128) {
    int tokl = c / 12;
    int r = c % 12;
    int mat = r >> 2, part = r & 3;
    size_t g = tok_g(tokl) * QKVC + mat * DIMC + head * 32 + part * 8;
    uint4 val = *reinterpret_cast<const uint4*>(&qkv[g]);
    _Float16* dst = (mat == 0) ? qS : (mat == 1) ? kS : vS;
    *reinterpret_cast<uint4*>(&dst[tokl * 32 + part * 8]) = val;
  }
  __syncthreads();

  // scores = q @ k^T  (16 tiles of 16x16, K=32; wave wv owns row-tile wv)
  v16h aq = ld_frag_row(&qS[(wv * 16 + ln) * 32], kb);
#pragma unroll
  for (int nt = 0; nt < 4; ++nt) {
    v16h bk = ld_frag_row(&kS[(nt * 16 + ln) * 32], kb);
    v8f acc = {};
    acc = wmma_f16(aq, bk, acc);
#pragma unroll
    for (int r = 0; r < 8; ++r) {
      int m = wv * 16 + ((lane < 16) ? r : 8 + r);
      sS[m * 64 + nt * 16 + ln] = acc[r];
    }
  }
  __syncthreads();

  // softmax over rows (scale 1/sqrt(32))
  if (t < 64) {
    float mx = -1e30f;
#pragma unroll 4
    for (int n = 0; n < 64; ++n) mx = fmaxf(mx, sS[t * 64 + n]);
    float s = 0.f;
#pragma unroll 4
    for (int n = 0; n < 64; ++n) {
      float e = expf((sS[t * 64 + n] - mx) * 0.17677669529663687f);
      sS[t * 64 + n] = e;
      s += e;
    }
    float inv = 1.0f / s;
#pragma unroll 4
    for (int n = 0; n < 64; ++n)
      pS[t * 64 + n] = (_Float16)(sS[t * 64 + n] * inv);
  }
  __syncthreads();

  // o = p @ v (8 tiles; wave wv owns row-tile wv, nt in {0,1}; K=64 -> 2 WMMA)
#pragma unroll
  for (int nt = 0; nt < 2; ++nt) {
    v8f acc = {};
#pragma unroll
    for (int kc = 0; kc < 2; ++kc) {
      v16h ap, bv;
#pragma unroll
      for (int j = 0; j < 8; ++j) {
        ap[j]     = pS[(wv * 16 + ln) * 64 + kc * 32 + kb + j];
        ap[j + 8] = pS[(wv * 16 + ln) * 64 + kc * 32 + kb + 16 + j];
        bv[j]     = vS[(kc * 32 + kb + j) * 32 + nt * 16 + ln];
        bv[j + 8] = vS[(kc * 32 + kb + 16 + j) * 32 + nt * 16 + ln];
      }
      acc = wmma_f16(ap, bv, acc);
    }
#pragma unroll
    for (int r = 0; r < 8; ++r) {
      int m = wv * 16 + ((lane < 16) ? r : 8 + r);
      obuf[tok_g(m) * DIMC + head * 32 + nt * 16 + ln] = (_Float16)acc[r];
    }
  }
}

// ============================ launcher ============================
extern "C" void kernel_launch(void* const* d_in, const int* in_sizes, int n_in,
                              void* d_out, int out_size, void* d_ws,
                              size_t ws_size, hipStream_t stream) {
  const float* x        = (const float*)d_in[0];
  const float* cgb_g    = (const float*)d_in[1];
  const float* cgb_b    = (const float*)d_in[2];
  const float* pw1_w    = (const float*)d_in[3];
  const float* pw1_b    = (const float*)d_in[4];
  const float* dw_w     = (const float*)d_in[5];
  const float* dw_b     = (const float*)d_in[6];
  const float* pw2_w    = (const float*)d_in[7];
  const float* pw2_b    = (const float*)d_in[8];
  const float* beta     = (const float*)d_in[9];
  const float* attn_g   = (const float*)d_in[10];
  const float* attn_b   = (const float*)d_in[11];
  const float* qkv_w    = (const float*)d_in[12];
  const float* qkv_b    = (const float*)d_in[13];
  const float* proj_w   = (const float*)d_in[14];
  const float* proj_b   = (const float*)d_in[15];
  const float* delta    = (const float*)d_in[16];
  const float* ffn_g    = (const float*)d_in[17];
  const float* ffn_b    = (const float*)d_in[18];
  const float* fc1_w    = (const float*)d_in[19];
  const float* fc1_b    = (const float*)d_in[20];
  const float* fc2_w    = (const float*)d_in[21];
  const float* fc2_b    = (const float*)d_in[22];
  const float* gamma    = (const float*)d_in[23];
  float* out = (float*)d_out;

  char* ws = (char*)d_ws;
  float*    master = (float*)(ws + WS_MASTER);
  _Float16* normed = (_Float16*)(ws + WS_NORMED);
  _Float16* big    = (_Float16*)(ws + WS_BIG);
  _Float16* gated  = (_Float16*)(ws + WS_GATED);
  float*    stats  = (float*)(ws + WS_STATS);
  _Float16* wf     = (_Float16*)(ws + WS_WF16);

  const dim3 blk256(256), blk128(128);

  // stats init + weight conversion
  k_init_stats<<<1, 32, 0, stream>>>(stats);
  k_f32_to_f16<<<(147456 + 255) / 256, blk256, 0, stream>>>(pw1_w,  wf + WF_PW1,  147456);
  k_f32_to_f16<<<(73728  + 255) / 256, blk256, 0, stream>>>(pw2_w,  wf + WF_PW2,  73728);
  k_f32_to_f16<<<(110592 + 255) / 256, blk256, 0, stream>>>(qkv_w,  wf + WF_QKV,  110592);
  k_f32_to_f16<<<(36864  + 255) / 256, blk256, 0, stream>>>(proj_w, wf + WF_PROJ, 36864);
  k_f32_to_f16<<<(147456 + 255) / 256, blk256, 0, stream>>>(fc1_w,  wf + WF_FC1,  147456);
  k_f32_to_f16<<<(73728  + 255) / 256, blk256, 0, stream>>>(fc2_w,  wf + WF_FC2,  73728);

  // x -> master (NHWC f32)
  k_nchw_to_nhwc<<<196608, blk256, 0, stream>>>(x, master);

  // ---- ConvGatedBlock ----
  k_gn_stats<<<dim3(512, BATCH), blk256, 0, stream>>>(master, stats + 0);
  k_gn_apply<<<49152, blk256, 0, stream>>>(master, stats + 0, cgb_g, cgb_b, normed);
  k_gemm<0, 4><<<dim3(MTOK / 64, B2HID / 128), blk256, 0, stream>>>(
      normed, wf + WF_PW1, pw1_b, B2HID, DIMC, big, nullptr, nullptr, nullptr);
  k_dwconv_gate<<<49152, blk256, 0, stream>>>(big, dw_w, dw_b, gated);
  k_gemm<1, 3><<<dim3(MTOK / 64, DIMC / 96), blk256, 0, stream>>>(
      gated, wf + WF_PW2, pw2_b, DIMC, HIDC, nullptr, beta, master, nullptr);

  // ---- shifted window attention ----
  k_gn_stats<<<dim3(512, BATCH), blk256, 0, stream>>>(master, stats + 8);
  k_gn_apply<<<49152, blk256, 0, stream>>>(master, stats + 8, attn_g, attn_b, normed);
  k_gemm<0, 3><<<dim3(MTOK / 64, QKVC / 96), blk256, 0, stream>>>(
      normed, wf + WF_QKV, qkv_b, QKVC, DIMC, big, nullptr, nullptr, nullptr);
  k_attn<<<dim3(4096, 6), blk128, 0, stream>>>(big, gated);
  k_gemm<1, 3><<<dim3(MTOK / 64, DIMC / 96), blk256, 0, stream>>>(
      gated, wf + WF_PROJ, proj_b, DIMC, DIMC, nullptr, delta, master, nullptr);

  // ---- GatedFFN ----
  k_gn_stats<<<dim3(512, BATCH), blk256, 0, stream>>>(master, stats + 16);
  k_gn_apply<<<49152, blk256, 0, stream>>>(master, stats + 16, ffn_g, ffn_b, normed);
  k_gemm<0, 4><<<dim3(MTOK / 64, B2HID / 128), blk256, 0, stream>>>(
      normed, wf + WF_FC1, fc1_b, B2HID, DIMC, big, nullptr, nullptr, nullptr);
  k_gate<<<49152, blk256, 0, stream>>>(big, gated);
  k_gemm<2, 3><<<dim3(MTOK / 64, DIMC / 96), blk256, 0, stream>>>(
      gated, wf + WF_FC2, fc2_b, DIMC, HIDC, nullptr, gamma, master, out);
}
